// HyperdimensionalEmbedding_67216238182702
// MI455X (gfx1250) — matrix operand order repaired and checked by reading, not compiled, not run on hardware
//
#include <hip/hip_runtime.h>

// ---------------------------------------------------------------------------
// Hyperdimensional embedding: gather -> circular conv (FFT) -> projection GEMM
//   B=4, S=4096, HD=4096, MODEL=512, VOCAB=50257
// Kernel 0: pack projection f32[4096][512] -> bf16 transposed [512][4096] (ws)
// Kernel 1: per-row packed FFT (DIF) -> Hermitian extract & multiply (bitrev
//           domain) -> inverse FFT (DIT) -> bound row stored bf16 in ws.
// Kernel 2: (16384x4096)bf16 @ (4096x512)bf16 -> f32, v_wmma_f32_16x16x32_bf16
//           with GLOBAL_LOAD_ASYNC_TO_LDS double-buffered pipeline (ASYNCcnt).
// ---------------------------------------------------------------------------

#define HD 4096
#define LOG2HD 12
#define MODEL 512
#define NROWS 16384   // B*S

typedef __attribute__((ext_vector_type(16))) __bf16 v16bf;
typedef __attribute__((ext_vector_type(8)))  float  v8f;

#if __has_builtin(__builtin_amdgcn_global_load_async_to_lds_b128) && \
    __has_builtin(__builtin_amdgcn_s_wait_asynccnt)
#define HAVE_ASYNC_LDS 1
#else
#define HAVE_ASYNC_LDS 0
#endif

// Builtin parameter types (from clang diagnostic): 16B int vector pointers,
// AS(1) for the global source, AS(3) for the LDS destination.
typedef int v4i __attribute__((vector_size(16)));
typedef __attribute__((address_space(1))) v4i gv4i;
typedef __attribute__((address_space(3))) v4i lv4i;

static __device__ __forceinline__ unsigned short f2bf(float f) {
    unsigned u = __builtin_bit_cast(unsigned, f);
    u += 0x7FFFu + ((u >> 16) & 1u);          // round-to-nearest-even
    return (unsigned short)(u >> 16);
}

// ====================== Kernel 0: pack projection matrix =====================
// P f32 [k=4096][n=512]  ->  Bp bf16 [n=512][k=4096]   (one-time, L2 resident)
__global__ __launch_bounds__(256) void hd_pack_b(
    const float* __restrict__ P, unsigned short* __restrict__ Bp)
{
    __shared__ __align__(16) float tile[64 * 65];   // [k][n], pad 65 vs banks

    const int tid = threadIdx.x;
    const int k0  = blockIdx.x * 64;     // 64 k-tiles
    const int n0  = blockIdx.y * 64;     // 8  n-tiles

    #pragma unroll
    for (int i = 0; i < 4; ++i) {        // 1024 float4 chunks
        int c  = tid + 256 * i;
        int r  = c >> 4;                 // k within tile
        int c4 = (c & 15) * 4;           // n within tile
        float4 v = *(const float4*)(P + (size_t)(k0 + r) * MODEL + n0 + c4);
        tile[r * 65 + c4 + 0] = v.x;
        tile[r * 65 + c4 + 1] = v.y;
        tile[r * 65 + c4 + 2] = v.z;
        tile[r * 65 + c4 + 3] = v.w;
    }
    __syncthreads();

    unsigned* out = (unsigned*)Bp;       // pack 2 bf16 per dword along k
    #pragma unroll
    for (int i = 0; i < 8; ++i) {        // 2048 dword chunks
        int c  = tid + 256 * i;
        int n  = c >> 5;                 // n within tile
        int kd = c & 31;                 // dword (2 k) within tile
        unsigned lo = (unsigned)f2bf(tile[(2 * kd + 0) * 65 + n]);
        unsigned hi = (unsigned)f2bf(tile[(2 * kd + 1) * 65 + n]);
        out[((size_t)(n0 + n) * HD + k0) / 2 + kd] = lo | (hi << 16);
    }
}

// ============================ Kernel 1: FFT bind ============================
__global__ __launch_bounds__(256) void hd_fft_bind(
    const int* __restrict__ token_ids,
    const float* __restrict__ base_vectors,
    const float* __restrict__ position_keys,
    unsigned short* __restrict__ bound_bf16)
{
    __shared__ __align__(16) float sre[HD];
    __shared__ __align__(16) float sim[HD];
    __shared__ __align__(16) float twr[HD / 2];
    __shared__ __align__(16) float twi[HD / 2];

    const int tid = threadIdx.x;
    const int row = blockIdx.x;                 // row = b*S + s
    const int s   = row & (HD - 1);             // S == 4096

    const int tok_id = token_ids[row];
    const float4* __restrict__ tokv = (const float4*)(base_vectors + (size_t)tok_id * HD);
    const float4* __restrict__ posv = (const float4*)(position_keys + (size_t)s * HD);

    #pragma unroll
    for (int i = 0; i < 4; ++i) {
        int j = tid + 256 * i;                  // 1024 float4 per array
        ((float4*)sre)[j] = tokv[j];
        ((float4*)sim)[j] = posv[j];
    }
    for (int j = tid; j < HD / 2; j += 256) {
        float sv, cv;
        __sincosf(-6.28318530717958647692f * (float)j * (1.0f / HD), &sv, &cv);
        twr[j] = cv; twi[j] = sv;
    }
    __syncthreads();

    // ---- Forward DIF FFT: natural in -> bit-reversed out ----
    #pragma unroll 1
    for (int hs = LOG2HD - 1; hs >= 0; --hs) {
        const int half = 1 << hs;
        #pragma unroll
        for (int q = tid; q < HD / 2; q += 256) {
            int j  = q & (half - 1);
            int i0 = ((q >> hs) << (hs + 1)) | j;
            int i1 = i0 + half;
            float ur = sre[i0], ui = sim[i0];
            float vr = sre[i1], vi = sim[i1];
            sre[i0] = ur + vr; sim[i0] = ui + vi;
            float dr = ur - vr, di = ui - vi;
            int ti = j << (LOG2HD - 1 - hs);
            float wr = twr[ti], wi = twi[ti];
            sre[i1] = dr * wr - di * wi;
            sim[i1] = dr * wi + di * wr;
        }
        __syncthreads();
    }

    // ---- Hermitian extract + spectral multiply (bit-reversed domain) ----
    const float scale = 1.0f / (float)HD;
    for (int k = tid; k <= HD / 2; k += 256) {   // only tid 0 reaches 2048
        if (k == 0) {
            float x = sre[0] * sim[0] * scale;
            sre[0] = x; sim[0] = 0.0f;
        } else if (k == HD / 2) {
            float x = sre[1] * sim[1] * scale;   // bitrev(2048)=1
            sre[1] = x; sim[1] = 0.0f;
        } else {
            int p1 = (int)(__brev((unsigned)k) >> (32 - LOG2HD));
            int p2 = (int)(__brev((unsigned)(HD - k)) >> (32 - LOG2HD));
            float a = sre[p1], b = sim[p1];
            float c = sre[p2], d = sim[p2];
            float tr = 0.5f * (a + c), ty = 0.5f * (b - d);
            float pr = 0.5f * (b + d), pi = 0.5f * (c - a);
            float xr = (tr * pr - ty * pi) * scale;
            float xi = (tr * pi + ty * pr) * scale;
            sre[p1] = xr;  sim[p1] = xi;
            sre[p2] = xr;  sim[p2] = -xi;
        }
    }
    __syncthreads();

    // ---- Inverse DIT FFT: bit-reversed in -> natural out (conj twiddles) ----
    #pragma unroll 1
    for (int hs = 0; hs < LOG2HD; ++hs) {
        const int half = 1 << hs;
        #pragma unroll
        for (int q = tid; q < HD / 2; q += 256) {
            int j  = q & (half - 1);
            int i0 = ((q >> hs) << (hs + 1)) | j;
            int i1 = i0 + half;
            int ti = j << (LOG2HD - 1 - hs);
            float wr = twr[ti], wi = -twi[ti];
            float vr = sre[i1], vi = sim[i1];
            float tr = vr * wr - vi * wi;
            float ty = vr * wi + vi * wr;
            float ur = sre[i0], ui = sim[i0];
            sre[i0] = ur + tr; sim[i0] = ui + ty;
            sre[i1] = ur - tr; sim[i1] = ui - ty;
        }
        __syncthreads();
    }

    unsigned* orow = (unsigned*)(bound_bf16 + (size_t)row * HD);
    #pragma unroll
    for (int i = 0; i < 8; ++i) {
        int j = tid + 256 * i;                   // 2048 dwords
        unsigned lo = (unsigned)f2bf(sre[2 * j]);
        unsigned hi = (unsigned)f2bf(sre[2 * j + 1]);
        orow[j] = lo | (hi << 16);
    }
}

// ============================ Kernel 2: WMMA GEMM ===========================
// C[16384x512] = A[16384x4096](bf16) * B(bf16, packed [n][k]), f32 accum.
// Tile: BM=128, BN=128, BK=64. 256 threads = 8 waves in 4(M)x2(N) grid.
// Async path: LDS double buffer, GLOBAL_LOAD_ASYNC_TO_LDS_B128, ASYNCcnt=8
// outstanding while tile kt's 16 WMMAs run — no VGPR staging, no spills.
#define BM 128
#define BN 128
#define BK 64
#define KTILES (HD / BK)
#define ASTR 72   // padded LDS strides (bf16 elems): 16B-aligned, bank-spread
#define BSTR 72
#if HAVE_ASYNC_LDS
#define NBUF 2
#else
#define NBUF 1
#endif
#define ABUF_ELEMS (BM * ASTR)
#define BBUF_ELEMS (BN * BSTR)
#define LDS_BYTES (NBUF * (ABUF_ELEMS + BBUF_ELEMS) * 2)

union FragBF { uint4 u[2]; v16bf v; };

__global__ __launch_bounds__(256) void hd_proj_gemm(
    const unsigned short* __restrict__ Abf,   // bound, row-major 16384x4096
    const unsigned short* __restrict__ Bp,    // packed projection [512][4096]
    float* __restrict__ C)                    // out, row-major 16384x512
{
    extern __shared__ __align__(16) unsigned short smem[];
    unsigned short* AsBase = smem;                       // NBUF * 128*72
    unsigned short* BsBase = smem + NBUF * ABUF_ELEMS;   // NBUF * 128*72

    const int tid   = threadIdx.x;
    const int lane  = tid & 31;
    const int wave  = tid >> 5;               // 0..7
    const int wm    = wave >> 1;              // 0..3 -> M offset wm*32
    const int wn    = wave & 1;               // 0..1 -> N offset wn*64
    const int h     = lane >> 4;              // lane half
    const int lm    = lane & 15;

    const int mb = (int)blockIdx.x >> 2;      // 128 M-blocks
    const int nb = (int)blockIdx.x & 3;       // 4 N-blocks

    const char* Agc = (const char*)(Abf + (size_t)(mb * BM) * HD);
    const char* Bgc = (const char*)(Bp + (size_t)(nb * BN) * HD);

    // staging chunk coords: 1024 chunks of 8 bf16 (16B), 4 per thread per mat
    unsigned goff[4];  int loff[4];
    #pragma unroll
    for (int i = 0; i < 4; ++i) {
        int c    = tid + 256 * i;
        int r    = c >> 3;
        int col  = (c & 7) * 8;
        goff[i]  = (unsigned)(r * HD + col) * 2u;   // global byte offset
        loff[i]  = r * ASTR + col;                  // LDS element offset
    }

    v8f acc[2][4];
    #pragma unroll
    for (int i = 0; i < 2; ++i)
        #pragma unroll
        for (int j = 0; j < 4; ++j) acc[i][j] = (v8f)0.0f;

#if HAVE_ASYNC_LDS
    auto stage_async = [&](int buf, int kt) {
        const char* ab = Agc + kt * (BK * 2);
        const char* bb = Bgc + kt * (BK * 2);
        unsigned short* Ad = AsBase + buf * ABUF_ELEMS;
        unsigned short* Bd = BsBase + buf * BBUF_ELEMS;
        #pragma unroll
        for (int i = 0; i < 4; ++i) {
            __builtin_amdgcn_global_load_async_to_lds_b128(
                (gv4i*)(ab + goff[i]), (lv4i*)(Ad + loff[i]), 0, 0);
            __builtin_amdgcn_global_load_async_to_lds_b128(
                (gv4i*)(bb + goff[i]), (lv4i*)(Bd + loff[i]), 0, 0);
        }
    };
#endif

    auto compute = [&](int buf) {
        const unsigned short* Ac = AsBase + buf * ABUF_ELEMS;
        const unsigned short* Bc = BsBase + buf * BBUF_ELEMS;
        #pragma unroll
        for (int kf = 0; kf < BK; kf += 32) {
            FragBF afr[2], bfr[4];
            #pragma unroll
            for (int fm = 0; fm < 2; ++fm) {
                const unsigned short* pa =
                    &Ac[(wm * 32 + fm * 16 + lm) * ASTR + kf + 8 * h];
                afr[fm].u[0] = *(const uint4*)pa;        // K = kf+8h   .. +7
                afr[fm].u[1] = *(const uint4*)(pa + 16); // K = kf+16+8h.. +7
            }
            #pragma unroll
            for (int fn = 0; fn < 4; ++fn) {
                const unsigned short* pb =
                    &Bc[(wn * 64 + fn * 16 + lm) * BSTR + kf + 8 * h];
                bfr[fn].u[0] = *(const uint4*)pb;        // K = kf+8h   .. +7
                bfr[fn].u[1] = *(const uint4*)(pb + 16); // K = kf+16+8h.. +7
            }
            #pragma unroll
            for (int fm = 0; fm < 2; ++fm)
                #pragma unroll
                for (int fn = 0; fn < 4; ++fn)
                    acc[fm][fn] = __builtin_amdgcn_wmma_f32_16x16x32_bf16(
                        false, afr[fm].v, false, bfr[fn].v,
                        (short)0, acc[fm][fn], false, false);
        }
    };

#if HAVE_ASYNC_LDS
    stage_async(0, 0);
    for (int kt = 0; kt < KTILES; ++kt) {
        const int cur = kt & 1;
        if (kt + 1 < KTILES) {
            stage_async(cur ^ 1, kt + 1);             // fills other buffer
            __builtin_amdgcn_s_wait_asynccnt(8);      // stage kt complete
        } else {
            __builtin_amdgcn_s_wait_asynccnt(0);
        }
        __syncthreads();                              // LDS visible to all
        compute(cur);
        __syncthreads();                              // reads done before reuse
    }
#else
    for (int kt = 0; kt < KTILES; ++kt) {
        uint4 ra[4], rb[4];
        const char* ab = Agc + kt * (BK * 2);
        const char* bb = Bgc + kt * (BK * 2);
        #pragma unroll
        for (int i = 0; i < 4; ++i) {
            ra[i] = *(const uint4*)(ab + goff[i]);
            rb[i] = *(const uint4*)(bb + goff[i]);
        }
        #pragma unroll
        for (int i = 0; i < 4; ++i) {
            *(uint4*)&AsBase[loff[i]] = ra[i];
            *(uint4*)&BsBase[loff[i]] = rb[i];
        }
        __syncthreads();
        compute(0);
        __syncthreads();
    }
#endif

    // ---- store C: c[j] maps to row (j + 8h), col lm of each 16x16 tile ----
    #pragma unroll
    for (int fm = 0; fm < 2; ++fm) {
        #pragma unroll
        for (int fn = 0; fn < 4; ++fn) {
            int rbase = mb * BM + wm * 32 + fm * 16 + 8 * h;
            int ccol  = nb * BN + wn * 64 + fn * 16 + lm;
            #pragma unroll
            for (int j = 0; j < 8; ++j)
                C[(size_t)(rbase + j) * MODEL + ccol] = acc[fm][fn][j];
        }
    }
}

// ================================ Launcher ==================================
extern "C" void kernel_launch(void* const* d_in, const int* in_sizes, int n_in,
                              void* d_out, int out_size, void* d_ws, size_t ws_size,
                              hipStream_t stream) {
    (void)in_sizes; (void)n_in; (void)out_size; (void)ws_size;
    const int*   token_ids     = (const int*)d_in[0];
    const float* base_vectors  = (const float*)d_in[1];
    const float* position_keys = (const float*)d_in[2];
    const float* projection    = (const float*)d_in[3];
    float* out = (float*)d_out;

    // workspace layout: [0,128MiB) bound bf16; [128MiB, +4MiB) packed B bf16
    unsigned short* bound = (unsigned short*)d_ws;
    unsigned short* Bpack = (unsigned short*)((char*)d_ws +
                             (size_t)NROWS * HD * sizeof(unsigned short));

    static int lds_cfg_done = 0;   // idempotent, deterministic
    if (!lds_cfg_done) {
        hipError_t e = hipFuncSetAttribute((const void*)hd_proj_gemm,
                            hipFuncAttributeMaxDynamicSharedMemorySize,
                            LDS_BYTES);
        (void)e;
        lds_cfg_done = 1;
    }

    hd_pack_b<<<dim3(HD / 64, MODEL / 64), 256, 0, stream>>>(projection, Bpack);
    hd_fft_bind<<<NROWS, 256, 0, stream>>>(token_ids, base_vectors,
                                           position_keys, bound);
    hd_proj_gemm<<<(NROWS / BM) * (MODEL / BN), 256, LDS_BYTES, stream>>>(
        bound, Bpack, out);
}
